// MixtureNet_10771777978612
// MI455X (gfx1250) — matrix operand order repaired and checked by reading, not compiled, not run on hardware
//
#include <hip/hip_runtime.h>
#include <hip/hip_bf16.h>
#include <math.h>

// ---------------------------------------------------------------------------
// MixtureNet forward for MI455X (gfx1250, wave32, WMMA).
// One 128-thread (4-wave) workgroup per batch element; all activations live in
// LDS; every matmul is v_wmma_f32_16x16x32_f16 (f16 in, f32 accumulate).
// Weights are pre-converted to f16 into d_ws so the B-operand gather is
// straight-line global_load_d16 with no per-element guards or cvts.
// ---------------------------------------------------------------------------

typedef __attribute__((ext_vector_type(16))) _Float16 v16h;
typedef __attribute__((ext_vector_type(8)))  float    v8f;

#define BATCH 4096
#define NTOK  20
#define NPAIR 190

// ---- flattened param-leaf indices (JAX pytree order: dict keys sorted,
// dense={b,w}, ln={b,g}, mab={f1,f2,ln1,ln2,wk,wo,wq,wv}, isab={ind,mab1,mab2})
enum {
  ACC1B = 0, ACC1W, ACC2B, ACC2W,
  COND1B, COND1W, COND2B, COND2W,
  CLN1B, CLN1G, CLN2B, CLN2G,
  FUS1B, FUS1W, FUS2B, FUS2W,
  FLN1B, FLN1G, FLN2B, FLN2G,
  HED1B, HED1W, HED2B, HED2W,
  ING1B, ING1W, ING2B, ING2W,
  ILN1B, ILN1G, ILN2B, ILN2G,
  ISAB1_IND  = 32,
  ISAB1_MAB1 = 33,   // 16 leaves
  ISAB1_MAB2 = 49,
  ISAB2_IND  = 65,
  ISAB2_MAB1 = 66,
  ISAB2_MAB2 = 82,
  MIX1B = 98, MIX1W = 99, MIX2B = 100, MIX2W = 101,
  PAIR1B = 102, PAIR1W = 103, PAIR2B = 104, PAIR2W = 105, PAIR3B = 106, PAIR3W = 107,
  PPROJB = 108, PPROJW = 109,
  PMA_MAB = 110,     // 16 leaves
  PMA_SEEDS = 126,
  SYN1B = 127, SYN1W = 128, SYN2B = 129, SYN2W = 130,
  TMP1B = 131, TMP1W = 132, TMP2B = 133, TMP2W = 134,
  NW = 135
};
// offsets inside one MAB leaf block
enum { MF1B = 0, MF1W, MF2B, MF2W, MLN1B, MLN1G, MLN2B, MLN2G,
       MWKB, MWKW, MWOB, MWOW, MWQB, MWQW, MWVB, MWVW };

struct KArgs {
  const float*    w[NW];          // f32 originals (biases, LN params, heads)
  const _Float16* wh[NW];         // f16 copies in d_ws (GEMM B operands)
  const float* ingredients;       // [B,20,31] f32
  const unsigned char* mask;      // [B,20] bool (True = padding)
  const float* cond;              // [B,20] f32
  float* out;                     // [B,103] f32
};

struct ConvArgs {
  const float* src[NW];
  _Float16*    dst[NW];
  int          sz[NW];
};

// sVec scratch offsets (floats)
#define POOLED 0
#define PFEAT  128
#define CONDV  256
#define FUSED  384
#define SMEAN  512
#define HBUF   640
#define RCNT   768
#define PSUM   769
#define PCNT   770
#define PSC    771
#define PI_    772
#define PJ_    804
#define PV_    836

// ---------------------------------------------------------------------------
// weight f32 -> f16 conversion (runs once per launch, before main kernel)
// ---------------------------------------------------------------------------
__global__ __launch_bounds__(256)
void convert_weights(ConvArgs c) {
  const int stride = gridDim.x * blockDim.x;
  const int tid = blockIdx.x * blockDim.x + threadIdx.x;
  for (int l = 0; l < NW; ++l) {
    const float* s = c.src[l];
    _Float16*    d = c.dst[l];
    const int    n = c.sz[l];
    for (int i = tid; i < n; i += stride)
      d[i] = (_Float16)s[i];
  }
}

// ---------------------------------------------------------------------------
// WMMA fragment helpers (wave32; A/B in contiguous k order, C/D per CDNA5 ISA 7.12.2)
// ---------------------------------------------------------------------------
// A (16x32 f16): lane m (0..15) holds K in {0..15}; lane m+16 holds
// K in {16..31}. element e -> k = base + e.
__device__ inline v16h ldA(const float* A, int lda, int m0, int k0) {
  const int lane = threadIdx.x & 31;
  const int m  = m0 + (lane & 15);
  const int kb = k0 + ((lane & 16) ? 16 : 0);
  v16h r;
#pragma unroll
  for (int e = 0; e < 16; ++e) {
    const int k = kb + e;
    r[e] = (_Float16)A[m * lda + k];
  }
  return r;
}

// C/D (16x16 f32): VGPR r: lanes 0..15 -> M=r, lanes 16..31 -> M=r+8; N=lane&15.
__device__ inline void stC(float* C, int ldc, int m0, int n0, v8f acc) {
  const int lane = threadIdx.x & 31;
  const int n  = n0 + (lane & 15);
  const int mo = (lane & 16) ? 8 : 0;
#pragma unroll
  for (int r = 0; r < 8; ++r)
    C[(m0 + mo + r) * ldc + n] = acc[r];
}

// Compile-time-specialized GEMM: C[Mt*16 x Nt*16] = A(lds f32) @ W(f16 global).
// B (32x16, K x N): lanes 0..15 hold K=0..15 of column n, lanes 16..31 K=16..31.
// Tail guards fold away at compile time for full K-steps / full-width N.
template<int LDA, int LDB, int Mt, int Nt, int Ks, int KVAL, int NVAL, int LDC>
__device__ void gemmT(const float* A, const _Float16* W, float* C) {
  const int wid  = threadIdx.x >> 5;
  const int lane = threadIdx.x & 31;
  for (int t = wid; t < Mt * Nt; t += 4) {
    const int mt = (t / Nt) * 16, nt = (t % Nt) * 16;
    const int n  = nt + (lane & 15);
    v8f acc = {};
#pragma unroll
    for (int ks = 0; ks < Ks; ++ks) {
      v16h a = ldA(A, LDA, mt, ks * 32);
      const int kb = ks * 32 + ((lane & 16) ? 16 : 0);
      v16h b;
      if ((ks + 1) * 32 <= KVAL && NVAL >= 16) {   // folds after unroll
        const _Float16* p = W + (size_t)kb * LDB + n;
#pragma unroll
        for (int e = 0; e < 16; ++e)
          b[e] = p[(size_t)e * LDB];
      } else {
#pragma unroll
        for (int e = 0; e < 16; ++e) {
          const int k = kb + e;
          b[e] = (k < KVAL && n < NVAL) ? W[(size_t)k * LDB + n]
                                        : (_Float16)0.f;
        }
      }
      acc = __builtin_amdgcn_wmma_f32_16x16x32_f16(false, a, false, b,
                                                   (short)0, acc, false, false);
    }
    stC(C, LDC, mt, nt, acc);
  }
}

// scores[h][q][k] = sum_d Q[q][h*32+d] * K[k][h*32+d]   (K-dim = head dim 32)
template<int MtQ, int NtK>
__device__ void gemm_scoresT(const float* Qs, const float* Ks_, float* Ss) {
  const int wid = threadIdx.x >> 5;
  constexpr int per = MtQ * NtK;
  for (int t = wid; t < 4 * per; t += 4) {
    const int hh = t / per, rem = t % per;
    const int mt = (rem / NtK) * 16, nt = (rem % NtK) * 16;
    v16h a = ldA(Qs, 128, mt, hh * 32);
    const int lane = threadIdx.x & 31;
    const int n  = nt + (lane & 15);
    const int kb = (lane & 16) ? 16 : 0;
    v16h b;
#pragma unroll
    for (int e = 0; e < 16; ++e)
      b[e] = (_Float16)Ks_[n * 128 + hh * 32 + kb + e];
    v8f acc = {};
    acc = __builtin_amdgcn_wmma_f32_16x16x32_f16(false, a, false, b,
                                                 (short)0, acc, false, false);
    stC(Ss + hh * 1024, 32, mt, nt, acc);
  }
}

// O[q][h*32+d] = sum_key S[h][q][key] * V[key][h*32+d]  (K-dim = 32 keys, padded)
template<int MtQ>
__device__ void gemm_attnvT(const float* Ss, const float* Vs, float* Os) {
  const int wid = threadIdx.x >> 5;
  constexpr int per = MtQ * 2;
  for (int t = wid; t < 4 * per; t += 4) {
    const int hh = t / per, rem = t % per;
    const int mt = (rem / 2) * 16, nt = (rem % 2) * 16;
    v16h a = ldA(Ss + hh * 1024, 32, mt, 0);
    const int lane = threadIdx.x & 31;
    const int n  = hh * 32 + nt + (lane & 15);
    const int kb = (lane & 16) ? 16 : 0;
    v16h b;
#pragma unroll
    for (int e = 0; e < 16; ++e)
      b[e] = (_Float16)Vs[(kb + e) * 128 + n];
    v8f acc = {};
    acc = __builtin_amdgcn_wmma_f32_16x16x32_f16(false, a, false, b,
                                                 (short)0, acc, false, false);
    stC(Os, 128, mt, hh * 32 + nt, acc);
  }
}

// ---------------------------------------------------------------------------
// elementwise helpers
// ---------------------------------------------------------------------------
__device__ inline float gelu_ex(float x) {
  return 0.5f * x * (1.f + erff(x * 0.7071067811865475f));
}

__device__ void bias_rows(float* B_, const float* bias, int rv, int rp, int dim) {
  for (int idx = threadIdx.x; idx < rp * dim; idx += blockDim.x) {
    const int r = idx / dim, d = idx - r * dim;
    B_[idx] = (r < rv) ? (B_[idx] + bias[d]) : 0.f;
  }
}

__device__ void bias_gelu_rows(float* B_, const float* bias, int rv, int rp, int dim) {
  for (int idx = threadIdx.x; idx < rp * dim; idx += blockDim.x) {
    const int r = idx / dim, d = idx - r * dim;
    B_[idx] = (r < rv) ? gelu_ex(B_[idx] + bias[d]) : 0.f;
  }
}

// dst = LN(src) * g + b  (eps 1e-5); rows >= rv zeroed; ld == dim (compact)
__device__ void lnorm_rows(const float* src, float* dst, const float* g,
                           const float* bb, int rv, int rp, int dim) {
  for (int r = threadIdx.x; r < rp; r += blockDim.x) {
    if (r < rv) {
      float mu = 0.f;
      for (int d = 0; d < dim; ++d) mu += src[r * dim + d];
      mu /= (float)dim;
      float var = 0.f;
      for (int d = 0; d < dim; ++d) { float t = src[r * dim + d] - mu; var += t * t; }
      var /= (float)dim;
      const float inv = rsqrtf(var + 1e-5f);
      for (int d = 0; d < dim; ++d)
        dst[r * dim + d] = (src[r * dim + d] - mu) * inv * g[d] + bb[d];
    } else {
      for (int d = 0; d < dim; ++d) dst[r * dim + d] = 0.f;
    }
  }
}

// ---------------------------------------------------------------------------
// MAB: Xq <- LN2(LN1(Xq + MHA(Xq,Xk)) + FFN(...)).  Xq/Xk are LDS [*,128].
// Shape-specialized: <NQV,NQP,NKV>.
// ---------------------------------------------------------------------------
template<int NQV, int NQP, int NKV>
__device__ void mab_apply(const float* const* w, const _Float16* const* wh,
                          float* Xq, const float* Xk, const unsigned char* km,
                          float* sQ, float* sK, float* sV, float* sS,
                          float* sA, float* sT) {
  constexpr int NKP = (NKV + 15) & ~15;
  constexpr int MtQ = NQP / 16, MtK = NKP / 16;

  gemmT<128,128,MtQ,8,4,128,128,128>(Xq, wh[MWQW], sQ);
  gemmT<128,128,MtK,8,4,128,128,128>(Xk, wh[MWKW], sK);
  gemmT<128,128,MtK,8,4,128,128,128>(Xk, wh[MWVW], sV);
  __syncthreads();
  bias_rows(sQ, w[MWQB], NQV, NQP, 128);
  bias_rows(sK, w[MWKB], NKV, NKP, 128);
  bias_rows(sV, w[MWVB], NKV, NKP, 128);
  __syncthreads();

  gemm_scoresT<MtQ, MtK>(sQ, sK, sS);
  __syncthreads();

  // masked softmax over keys (scale 1/sqrt(32)); zero weights for padded keys
  for (int idx = threadIdx.x; idx < 4 * NQV; idx += blockDim.x) {
    const int hh = idx / NQV, q = idx % NQV;
    float* row = sS + hh * 1024 + q * 32;
    float mx = -1e30f;
    for (int k = 0; k < NKV; ++k) {
      float s = row[k] * 0.17677669529663687f;
      if (km && km[k]) s = -1e9f;
      row[k] = s;
      mx = fmaxf(mx, s);
    }
    float sum = 0.f;
    for (int k = 0; k < NKV; ++k) { float e = expf(row[k] - mx); row[k] = e; sum += e; }
    const float inv = 1.f / sum;
    for (int k = 0; k < NKV; ++k) row[k] *= inv;
    for (int k = NKV; k < 32; ++k) row[k] = 0.f;
  }
  __syncthreads();

  gemm_attnvT<MtQ>(sS, sV, sT);
  __syncthreads();
  gemmT<128,128,MtQ,8,4,128,128,128>(sT, wh[MWOW], sA);  // output proj
  __syncthreads();
  for (int idx = threadIdx.x; idx < NQP * 128; idx += blockDim.x) {
    const int r = idx >> 7, d = idx & 127;
    sA[idx] = (r < NQV) ? (sA[idx] + w[MWOB][d] + Xq[idx]) : 0.f;
  }
  __syncthreads();
  lnorm_rows(sA, Xq, w[MLN1G], w[MLN1B], NQV, NQP, 128);
  __syncthreads();

  // FFN 128 -> 256 -> 128
  gemmT<128,256,MtQ,16,4,128,256,256>(Xq, wh[MF1W], sT);
  __syncthreads();
  bias_gelu_rows(sT, w[MF1B], NQV, NQP, 256);
  __syncthreads();
  gemmT<256,128,MtQ,8,8,256,128,128>(sT, wh[MF2W], sA);
  __syncthreads();
  for (int idx = threadIdx.x; idx < NQP * 128; idx += blockDim.x) {
    const int r = idx >> 7, d = idx & 127;
    sA[idx] = (r < NQV) ? (sA[idx] + w[MF2B][d] + Xq[idx]) : 0.f;
  }
  __syncthreads();
  lnorm_rows(sA, Xq, w[MLN2G], w[MLN2B], NQV, NQP, 128);
  __syncthreads();
}

// small dense head: out = act2(W2 @ gelu(W1 @ x + b1) + b2)
__device__ void head2(const float* x, const float* w1, const float* b1, int h,
                      const float* w2, const float* b2, int o,
                      int act, float* outp, float* hbuf) {
  for (int j = threadIdx.x; j < h; j += blockDim.x) {
    float a = b1[j];
    for (int k = 0; k < 128; ++k) a += x[k] * w1[k * h + j];
    hbuf[j] = gelu_ex(a);
  }
  __syncthreads();
  for (int j = threadIdx.x; j < o; j += blockDim.x) {
    float a = b2[j];
    for (int k = 0; k < h; ++k) a += hbuf[k] * w2[k * o + j];
    if (act == 0)      a = 1.f / (1.f + expf(-a));     // sigmoid
    else if (act == 2) a = tanhf(a);                   // tanh
    outp[j] = a;
  }
  __syncthreads();
}

// ---------------------------------------------------------------------------
__global__ __launch_bounds__(128)
void mixture_kernel(KArgs args) {
  extern __shared__ float smem[];
  float* sH   = smem;          // 32x128  = 4096
  float* sI   = sH + 4096;     // 16x128  = 2048
  float* sQ   = sI + 2048;     // 32x128
  float* sK   = sQ + 4096;     // 32x128
  float* sV   = sK + 4096;     // 32x128
  float* sS   = sV + 4096;     // 4x32x32 = 4096
  float* sA   = sS + 4096;     // 32x256  = 8192
  float* sT   = sA + 8192;     // 32x256  = 8192
  float* sVec = sT + 8192;     // 1024

  const int b = blockIdx.x;
  const unsigned char* km = args.mask + (size_t)b * NTOK;

  // warm L2/WGP$ with f16 weight bases (global_prefetch_b8)
  for (int i = threadIdx.x; i < NW; i += blockDim.x)
    __builtin_prefetch(args.wh[i], 0, 1);

  // ---- ingredient encoder: [20x31] -> 64 -> 128 -------------------------
  for (int idx = threadIdx.x; idx < 32 * 32; idx += blockDim.x) {
    const int r = idx >> 5, c = idx & 31;
    sA[idx] = (r < NTOK && c < 31)
              ? args.ingredients[((size_t)b * NTOK + r) * 31 + c] : 0.f;
  }
  __syncthreads();
  gemmT<32,64,2,4,1,31,64,64>(sA, args.wh[ING1W], sT);
  __syncthreads();
  bias_gelu_rows(sT, args.w[ING1B], NTOK, 32, 64);
  __syncthreads();
  lnorm_rows(sT, sA, args.w[ILN1G], args.w[ILN1B], NTOK, 32, 64);
  __syncthreads();
  gemmT<64,128,2,8,2,64,128,128>(sA, args.wh[ING2W], sT);
  __syncthreads();
  bias_gelu_rows(sT, args.w[ING2B], NTOK, 32, 128);
  __syncthreads();
  lnorm_rows(sT, sH, args.w[ILN2G], args.w[ILN2B], NTOK, 32, 128);
  __syncthreads();

  // ---- ISAB x2 ----------------------------------------------------------
  for (int pass = 0; pass < 2; ++pass) {
    const int indIdx = pass ? ISAB2_IND : ISAB1_IND;
    const int m1 = pass ? ISAB2_MAB1 : ISAB1_MAB1;
    const int m2 = pass ? ISAB2_MAB2 : ISAB1_MAB2;
    for (int idx = threadIdx.x; idx < 16 * 128; idx += blockDim.x)
      sI[idx] = args.w[indIdx][idx];
    __syncthreads();
    mab_apply<16,16,NTOK>(args.w + m1, args.wh + m1, sI, sH, km,
                          sQ, sK, sV, sS, sA, sT);
    mab_apply<NTOK,32,16>(args.w + m2, args.wh + m2, sH, sI, nullptr,
                          sQ, sK, sV, sS, sA, sT);
  }

  // ---- PMA pooling -------------------------------------------------------
  for (int idx = threadIdx.x; idx < 16 * 128; idx += blockDim.x)
    sI[idx] = (idx < 128) ? args.w[PMA_SEEDS][idx] : 0.f;
  __syncthreads();
  mab_apply<1,16,NTOK>(args.w + PMA_MAB, args.wh + PMA_MAB, sI, sH, km,
                       sQ, sK, sV, sS, sA, sT);
  for (int d = threadIdx.x; d < 128; d += blockDim.x)
    sVec[POOLED + d] = sI[d];
  if (threadIdx.x == 0) {
    int rc = 0;
    for (int i = 0; i < NTOK; ++i) rc += km[i] ? 0 : 1;
    sVec[RCNT] = (float)rc;
    sVec[PSUM] = 0.f;
    sVec[PCNT] = 0.f;
  }
  __syncthreads();

  // set_mean over real tokens
  for (int d = threadIdx.x; d < 128; d += blockDim.x) {
    float s = 0.f;
    for (int i = 0; i < NTOK; ++i) if (!km[i]) s += sH[i * 128 + d];
    sVec[SMEAN + d] = s / fmaxf(sVec[RCNT], 1.f);
  }
  __syncthreads();

  // ---- pairwise edge-MLP over 190 pairs, 6 chunks of 32 ------------------
  for (int c = 0; c < 6; ++c) {
    for (int t = threadIdx.x; t < 32; t += blockDim.x) {
      const int p = c * 32 + t;
      int i = 0, j = 0, valid = 0;
      if (p < NPAIR) {
        int k = p;
        while (k >= NTOK - 1 - i) { k -= NTOK - 1 - i; ++i; }
        j = i + 1 + k;
        valid = (!km[i] && !km[j]) ? 1 : 0;
      }
      sVec[PI_ + t] = (float)i;
      sVec[PJ_ + t] = (float)j;
      sVec[PV_ + t] = (float)valid;
    }
    __syncthreads();
    for (int idx = threadIdx.x; idx < 32 * 256; idx += blockDim.x) {
      const int t = idx >> 8, d = idx & 255;
      const int p = c * 32 + t;
      float v = 0.f;
      if (p < NPAIR) {
        const int i = (int)sVec[PI_ + t], j = (int)sVec[PJ_ + t];
        v = (d < 128) ? sH[i * 128 + d] : sH[j * 128 + (d - 128)];
      }
      sA[idx] = v;
    }
    __syncthreads();
    gemmT<256,128,2,8,8,256,128,128>(sA, args.wh[PAIR1W], sT);
    __syncthreads();
    bias_gelu_rows(sT, args.w[PAIR1B], 32, 32, 128);
    __syncthreads();
    gemmT<128,64,2,4,4,128,64,64>(sT, args.wh[PAIR2W], sA);
    __syncthreads();
    bias_gelu_rows(sA, args.w[PAIR2B], 32, 32, 64);
    __syncthreads();
    gemmT<64,1,2,1,2,64,1,16>(sA, args.wh[PAIR3W], sT);
    __syncthreads();
    if (threadIdx.x == 0) {
      float s = sVec[PSUM], n = sVec[PCNT];
      for (int t = 0; t < 32; ++t) {
        const float sc = tanhf(sT[t * 16] + args.w[PAIR3B][0]);
        s += sc * sVec[PV_ + t];
        n += sVec[PV_ + t];
      }
      sVec[PSUM] = s;
      sVec[PCNT] = n;
    }
    __syncthreads();
  }
  if (threadIdx.x == 0)
    sVec[PSC] = sVec[PSUM] / fmaxf(sVec[PCNT], 1.f);
  __syncthreads();

  // ---- pair_proj([set_mean, pair_scores]) : K=129 padded to 160 ----------
  for (int idx = threadIdx.x; idx < 16 * 160; idx += blockDim.x) {
    const int r = idx / 160, d = idx - r * 160;
    float v = 0.f;
    if (r == 0) {
      if (d < 128)       v = sVec[SMEAN + d];
      else if (d == 128) v = sVec[PSC];
    }
    sA[idx] = v;
  }
  __syncthreads();
  gemmT<160,128,1,8,5,129,128,128>(sA, args.wh[PPROJW], sT);
  __syncthreads();
  for (int d = threadIdx.x; d < 128; d += blockDim.x)
    sVec[PFEAT + d] = sT[d] + args.w[PPROJB][d];
  __syncthreads();

  // ---- condition encoder: 20 -> 64 -> 128 --------------------------------
  for (int idx = threadIdx.x; idx < 16 * 32; idx += blockDim.x) {
    const int r = idx >> 5, d = idx & 31;
    sA[idx] = (r == 0 && d < 20) ? args.cond[(size_t)b * 20 + d] : 0.f;
  }
  __syncthreads();
  gemmT<32,64,1,4,1,20,64,64>(sA, args.wh[COND1W], sT);
  __syncthreads();
  bias_gelu_rows(sT, args.w[COND1B], 1, 16, 64);
  __syncthreads();
  lnorm_rows(sT, sA, args.w[CLN1G], args.w[CLN1B], 1, 16, 64);
  __syncthreads();
  gemmT<64,128,1,8,2,64,128,128>(sA, args.wh[COND2W], sT);
  __syncthreads();
  bias_gelu_rows(sT, args.w[COND2B], 1, 16, 128);
  __syncthreads();
  lnorm_rows(sT, sA, args.w[CLN2G], args.w[CLN2B], 1, 16, 128);
  __syncthreads();
  for (int d = threadIdx.x; d < 128; d += blockDim.x)
    sVec[CONDV + d] = sA[d];
  __syncthreads();

  // ---- fusion: 384 -> 256 -> 128 -----------------------------------------
  for (int idx = threadIdx.x; idx < 16 * 384; idx += blockDim.x) {
    const int r = idx / 384, d = idx - r * 384;
    float v = 0.f;
    if (r == 0)
      v = (d < 128) ? sVec[POOLED + d]
        : (d < 256) ? sVec[PFEAT + (d - 128)]
                    : sVec[CONDV + (d - 256)];
    sA[idx] = v;
  }
  __syncthreads();
  gemmT<384,256,1,16,12,384,256,256>(sA, args.wh[FUS1W], sT);
  __syncthreads();
  bias_gelu_rows(sT, args.w[FUS1B], 1, 16, 256);
  __syncthreads();
  lnorm_rows(sT, sA, args.w[FLN1G], args.w[FLN1B], 1, 16, 256);
  __syncthreads();
  gemmT<256,128,1,8,8,256,128,128>(sA, args.wh[FUS2W], sT);
  __syncthreads();
  bias_gelu_rows(sT, args.w[FUS2B], 1, 16, 128);
  __syncthreads();
  lnorm_rows(sT, sA, args.w[FLN2G], args.w[FLN2B], 1, 16, 128);
  __syncthreads();
  for (int d = threadIdx.x; d < 128; d += blockDim.x)
    sVec[FUSED + d] = sA[d];
  __syncthreads();

  // ---- heads: [mix 22 | tmp 66 | syn 1 | acc 12 | hed 1 | pair 1] = 103 --
  float* outb = args.out + (size_t)b * 103;
  const float* fz = sVec + FUSED;
  head2(fz, args.w[MIX1W], args.w[MIX1B], 64,  args.w[MIX2W], args.w[MIX2B], 22, 0, outb + 0,  sVec + HBUF);
  head2(fz, args.w[TMP1W], args.w[TMP1B], 128, args.w[TMP2W], args.w[TMP2B], 66, 0, outb + 22, sVec + HBUF);
  head2(fz, args.w[SYN1W], args.w[SYN1B], 32,  args.w[SYN2W], args.w[SYN2B], 1,  0, outb + 88, sVec + HBUF);
  head2(fz, args.w[ACC1W], args.w[ACC1B], 32,  args.w[ACC2W], args.w[ACC2B], 12, 1, outb + 89, sVec + HBUF);
  head2(fz, args.w[HED1W], args.w[HED1B], 32,  args.w[HED2W], args.w[HED2B], 1,  2, outb + 101, sVec + HBUF);
  if (threadIdx.x == 0) outb[102] = sVec[PSC];
}

// ---------------------------------------------------------------------------
extern "C" void kernel_launch(void* const* d_in, const int* in_sizes, int n_in,
                              void* d_out, int out_size, void* d_ws, size_t ws_size,
                              hipStream_t stream) {
  (void)n_in; (void)out_size; (void)ws_size;
  KArgs a;
  ConvArgs c;
  size_t off = 0;                      // in f16 elements, 16-aligned per leaf
  for (int i = 0; i < NW; ++i) {
    a.w[i]   = (const float*)d_in[i];
    c.src[i] = (const float*)d_in[i];
    c.sz[i]  = in_sizes[i];
    _Float16* p = (_Float16*)d_ws + off;
    c.dst[i] = p;
    a.wh[i]  = p;
    off += (size_t)((in_sizes[i] + 15) & ~15);
  }
  a.ingredients = (const float*)d_in[NW + 0];         // 135
  a.mask        = (const unsigned char*)d_in[NW + 1]; // 136 (bool array)
  a.cond        = (const float*)d_in[NW + 2];         // 137
  a.out         = (float*)d_out;

  // 1) convert all weight leaves f32 -> f16 into d_ws (~1.7 MB)
  convert_weights<<<dim3(512), dim3(256), 0, stream>>>(c);
  // 2) main fused forward
  const size_t shbytes = (size_t)39936 * sizeof(float); // ~156 KB LDS / WG
  mixture_kernel<<<dim3(BATCH), dim3(128), shbytes, stream>>>(a);
}
